// self_attention_51702816309745
// MI455X (gfx1250) — matrix operand order repaired and checked by reading, not compiled, not run on hardware
//
#include <hip/hip_runtime.h>
#include <hip/hip_bf16.h>
#include <stdint.h>

// Problem constants (match reference).
#define BATCH  8
#define SEQ    2048
#define DMODEL 2048
#define DHEAD  128
#define NCHUNK (DMODEL / 32)   // 64 K-chunks in the projection GEMM

// LDS tile geometry for the flash kernel (TDM-padded rows; see pad fields).
#define KROW_STRIDE 272                    // 256B data + 16B pad per key row
#define KTILE_BYTES (32 * KROW_STRIDE)     // 8704B
#define VROW_STRIDE 80                     // 64B data + 16B pad per vT row
#define VTILE_BYTES (128 * VROW_STRIDE)    // 10240B

typedef __attribute__((ext_vector_type(16))) __bf16 v16bf;
typedef __attribute__((ext_vector_type(8)))  float  v8f;
typedef __attribute__((ext_vector_type(4)))  unsigned int v4u;
typedef __attribute__((ext_vector_type(8)))  int v8i;
typedef __attribute__((ext_vector_type(4)))  int v4i;

// 32-byte fragment viewed both as two 16B payloads and as a WMMA operand.
union Frag16 {
  uint4 u[2];
  v16bf v;
};

// f32 -> bf16 via the hardware convert (v_cvt_pk_bf16_f32 pairs).
__device__ __forceinline__ __bf16 f2bf(float f) { return (__bf16)f; }
__device__ __forceinline__ unsigned short f2bf_bits(float f) {
  __bf16 h = (__bf16)f;
  unsigned short u;
  __builtin_memcpy(&u, &h, 2);
  return u;
}

__device__ __forceinline__ v8f wmma_bf16(v16bf a, v16bf b, v8f c) {
  return __builtin_amdgcn_wmma_f32_16x16x32_bf16(
      false, a, false, b, (short)0, c, false, false);
}

// LDS byte offset of a __shared__ pointer (generic-pointer low 32 bits).
__device__ __forceinline__ unsigned int lds_off(const void* p) {
  return (unsigned int)(uintptr_t)p;
}

// ---------------------------------------------------------------------------
// Tensor Data Mover: 2D tile load global->LDS (D# per CDNA5 ISA ch.8).
// tile_w elements (2B each) per row, tile_h rows, row stride in elements.
// pad_interval/pad_amount per D# encoding (pad_en=0 -> padding disabled).
// 6-arg builtin form: (g0, g1, g2, g3, g4-spare, cpol).
// ---------------------------------------------------------------------------
__device__ __forceinline__ void tdm_load_2d(unsigned int ldsaddr, const void* gptr,
                                            unsigned int tile_w, unsigned int tile_h,
                                            unsigned int row_stride,
                                            unsigned int pad_interval,
                                            unsigned int pad_amount,
                                            unsigned int pad_en) {
  const unsigned long long ga = (unsigned long long)gptr;
  v4u g0;
  g0[0] = 1u;                                    // count=1 (valid), user mode
  g0[1] = ldsaddr;                               // lds_addr
  g0[2] = (unsigned int)ga;                      // global_addr[31:0]
  g0[3] = (unsigned int)((ga >> 32) & 0x01FFFFFFu) | (2u << 30);  // addr[56:32], type=2

  const unsigned int td0 = 1u << 20;             // tensor_dim0: huge (no OOB clip)
  const unsigned int td1 = 1u << 20;             // tensor_dim1: huge
  v8i g1;
  g1[0] = (int)((1u << 16)                       // data_size = 1 (2 bytes)
                | (pad_en << 20)                 // pad_enable
                | (pad_interval << 22)           // pad interval code
                | (pad_amount << 25));           // pad amount code
  g1[1] = (int)((td0 & 0xFFFFu) << 16);          // [63:48] tensor_dim0 lo
  g1[2] = (int)((td0 >> 16) | ((td1 & 0xFFFFu) << 16));  // td0 hi | td1 lo
  g1[3] = (int)((td1 >> 16) | (tile_w << 16));   // td1 hi | tile_dim0
  g1[4] = (int)(tile_h);                         // tile_dim1 | tile_dim2=0
  g1[5] = (int)row_stride;                       // tensor_dim0_stride lo
  g1[6] = 0;                                     // stride hi | dim1_stride lo
  g1[7] = 0;
  const v4i z4 = {0, 0, 0, 0};                   // groups 2/3: unused (2D tensor)
  const v8i z8 = {0, 0, 0, 0, 0, 0, 0, 0};       // spare descriptor group
  __builtin_amdgcn_tensor_load_to_lds(g0, g1, z4, z4, z8, 0);
}

// ---------------------------------------------------------------------------
// Kernel 1: pack weights -> bf16 in fragment/plane order:
//   16B-group index = ((widx*64 + c)*8 + nt)*2 + plane, then [lane 0..31].
// ---------------------------------------------------------------------------
__global__ void pack_w_kernel(const float* __restrict__ Wq,
                              const float* __restrict__ Wk,
                              const float* __restrict__ Wv,
                              unsigned short* __restrict__ Wtf) {
  int gid = blockIdx.x * blockDim.x + threadIdx.x;
  const int ngroups = 3 * NCHUNK * 8 * 2 * 32;   // 98304
  if (gid >= ngroups) return;
  const int lane  = gid & 31;
  const int plane = (gid >> 5) & 1;
  const int nt    = (gid >> 6) & 7;
  const int c     = (gid >> 9) & 63;
  const int widx  = gid >> 15;
  const int lhalf = lane & 15;
  const bool hi   = lane >= 16;

  const int n  = nt * 16 + lhalf;
  const int k0 = c * 32 + (hi ? 8 : 0) + plane * 16;
  const float* W = (widx == 0) ? Wq : (widx == 1) ? Wk : Wv;

  unsigned int d[4];
#pragma unroll
  for (int j = 0; j < 4; ++j) {
    unsigned int lo = f2bf_bits(W[(size_t)(k0 + 2 * j) * DHEAD + n]);
    unsigned int hb = f2bf_bits(W[(size_t)(k0 + 2 * j + 1) * DHEAD + n]);
    d[j] = lo | (hb << 16);
  }
  *(uint4*)(Wtf + (size_t)gid * 8) = make_uint4(d[0], d[1], d[2], d[3]);
}

// ---------------------------------------------------------------------------
// Kernel 2: QKV projection GEMM; B chunks staged by the TDM (8KB flat tiles),
// double-buffered LDS, one barrier per K-chunk, A registers ping-ponged.
// ---------------------------------------------------------------------------
__device__ __forceinline__ void gemm_load_a(float4 f[4], const float* __restrict__ xrow,
                                            int k0, bool hi) {
  const int ka = k0 + (hi ? 8 : 0);
  f[0] = *(const float4*)(xrow + ka);
  f[1] = *(const float4*)(xrow + ka + 4);
  f[2] = *(const float4*)(xrow + ka + 16);
  f[3] = *(const float4*)(xrow + ka + 20);
}

__device__ __forceinline__ void gemm_consume(v8f acc[8], const float4 f[4],
                                             const uint4 (*buf)[2][32], int lane) {
  Frag16 bf[8];
#pragma unroll
  for (int nt = 0; nt < 8; ++nt) {
    bf[nt].u[0] = buf[nt][0][lane];
    bf[nt].u[1] = buf[nt][1][lane];
  }
  v16bf a;
#pragma unroll
  for (int i = 0; i < 4; ++i) {
    a[4 * i + 0] = f2bf(f[i].x);
    a[4 * i + 1] = f2bf(f[i].y);
    a[4 * i + 2] = f2bf(f[i].z);
    a[4 * i + 3] = f2bf(f[i].w);
  }
#pragma unroll
  for (int nt = 0; nt < 8; ++nt) acc[nt] = wmma_bf16(a, bf[nt].v, acc[nt]);
}

__global__ void __launch_bounds__(256)
qkv_gemm_kernel(const float* __restrict__ x,
                const unsigned short* __restrict__ Wtf,
                const float* __restrict__ bq,
                const float* __restrict__ bk,
                const float* __restrict__ bv,
                unsigned short* __restrict__ qb,
                unsigned short* __restrict__ kbf,
                unsigned short* __restrict__ vTb) {
  __shared__ __align__(16) uint4 ldsB[2][8][2][32];  // [buf][frag][plane][lane], 16KB

  const int lane  = threadIdx.x & 31;
  const int wave  = threadIdx.x >> 5;
  const int lhalf = lane & 15;
  const bool hi   = lane >= 16;
  const int widx  = blockIdx.y;
  const int mw    = blockIdx.x * 128 + wave * 16;

  const unsigned short* Wf = Wtf + (size_t)widx * NCHUNK * 4096;  // 4096 bf16 / chunk
  const float* bias = (widx == 0) ? bq : (widx == 1) ? bk : bv;

  v8f acc[8];
#pragma unroll
  for (int i = 0; i < 8; ++i)
#pragma unroll
    for (int j = 0; j < 8; ++j) acc[i][j] = 0.0f;

  const float* xrow = x + (size_t)(mw + lhalf) * DMODEL;

  // TDM stage: one flat 8KB tile per chunk, issued by wave 0 only.
  auto stage = [&](int c, int buf) {
    tdm_load_2d(lds_off(&ldsB[buf][0][0][0]), Wf + (size_t)c * 4096,
                /*tile_w=*/4096, /*tile_h=*/1, /*row_stride=*/4096,
                /*pad_interval=*/0, /*pad_amount=*/0, /*pad_en=*/0);
  };

  float4 fA[4], fB[4];
  if (wave == 0) {
    stage(0, 0);
    __builtin_amdgcn_s_wait_tensorcnt(0);
  }
  gemm_load_a(fA, xrow, 0, hi);
  __syncthreads();

  for (int c = 0; c < NCHUNK; c += 2) {
    if (c + 1 < NCHUNK) {
      gemm_load_a(fB, xrow, (c + 1) * 32, hi);
      if (wave == 0) stage(c + 1, 1);
    }
    gemm_consume(acc, fA, ldsB[0], lane);
    if (wave == 0) __builtin_amdgcn_s_wait_tensorcnt(0);
    __syncthreads();

    if (c + 2 < NCHUNK) {
      gemm_load_a(fA, xrow, (c + 2) * 32, hi);
      if (wave == 0) stage(c + 2, 0);
    }
    gemm_consume(acc, fB, ldsB[1], lane);
    if (wave == 0) __builtin_amdgcn_s_wait_tensorcnt(0);
    __syncthreads();
  }

  // Epilogue.  C-layout: VGPR r, lanes 0-15 -> row mw+r; lanes 16-31 -> mw+8+r.
  const int bidx  = mw / SEQ;
  const int srow0 = mw % SEQ;
#pragma unroll
  for (int nt = 0; nt < 8; ++nt) {
    const int col = nt * 16 + lhalf;
    const float bs = bias[col];
    if (widx < 2) {
      unsigned short* dst = (widx == 0) ? qb : kbf;
#pragma unroll
      for (int r = 0; r < 8; ++r) {
        const int row = mw + (hi ? 8 + r : r);
        dst[(size_t)row * DHEAD + col] = f2bf_bits(acc[nt][r] + bs);
      }
    } else {
      // vT[b][col][s]: one lane's 8 acc entries are 8 consecutive s values.
      unsigned int p[4];
#pragma unroll
      for (int j = 0; j < 4; ++j) {
        unsigned int lo = f2bf_bits(acc[nt][2 * j] + bs);
        unsigned int hb = f2bf_bits(acc[nt][2 * j + 1] + bs);
        p[j] = lo | (hb << 16);
      }
      size_t off = ((size_t)bidx * DHEAD + col) * SEQ + srow0 + (hi ? 8 : 0);
      *(uint4*)(vTb + off) = make_uint4(p[0], p[1], p[2], p[3]);
    }
  }
}

// ---------------------------------------------------------------------------
// Kernel 3: flash attention; k/v tiles staged by the TDM with LDS row padding
// chosen so the 16-lane fragment reads are bank-conflict free.
//   k tile: 32 rows x 256B (+16B pad) ; v tile: 128 rows x 64B (+16B pad).
// ---------------------------------------------------------------------------
__global__ void __launch_bounds__(256)
flash_attn_kernel(const unsigned short* __restrict__ qb,
                  const unsigned short* __restrict__ kbf,
                  const unsigned short* __restrict__ vTb,
                  float* __restrict__ out) {
  __shared__ __align__(16) unsigned char ldsK[2][KTILE_BYTES];       // 17408B
  __shared__ __align__(16) unsigned char ldsV[2][VTILE_BYTES];       // 20480B
  __shared__ __align__(16) unsigned short ldsP[8][16 * 32];          // 8192B

  const int lane  = threadIdx.x & 31;
  const int wave  = threadIdx.x >> 5;
  const int lhalf = lane & 15;
  const bool hi   = lane >= 16;

  const int mw = blockIdx.x * 128 + wave * 16;
  const int b  = mw / SEQ;
  const int s0 = mw % SEQ;

  const unsigned short* kbase = kbf + (size_t)b * SEQ * DHEAD;
  const unsigned short* vbase = vTb + (size_t)b * DHEAD * SEQ;

  // q A-fragments: 4 chunks of K=32 along DHEAD, loaded once.
  v16bf qf[4];
  {
    const unsigned short* qrow = qb + ((size_t)b * SEQ + s0 + lhalf) * DHEAD;
#pragma unroll
    for (int dc = 0; dc < 4; ++dc) {
      const int d0 = dc * 32 + (hi ? 16 : 0);
      Frag16 f;
      f.u[0] = *(const uint4*)(qrow + d0);
      f.u[1] = *(const uint4*)(qrow + d0 + 8);
      qf[dc] = f.v;
    }
  }

  // Wave 0 stages both tiles for key chunk j0 via the TDM.
  auto stage_kv = [&](int j0, int buf) {
    // k: rows = keys j0..j0+31, 128 bf16 each; pad 4 dwords every 64 dwords.
    tdm_load_2d(lds_off(&ldsK[buf][0]), kbase + (size_t)j0 * DHEAD,
                /*tile_w=*/DHEAD, /*tile_h=*/32, /*row_stride=*/DHEAD,
                /*pad_interval=*/5, /*pad_amount=*/3, /*pad_en=*/1);
    // v: rows = head dims 0..127, 32 bf16 each; pad 4 dwords every 16 dwords.
    tdm_load_2d(lds_off(&ldsV[buf][0]), vbase + j0,
                /*tile_w=*/32, /*tile_h=*/DHEAD, /*row_stride=*/SEQ,
                /*pad_interval=*/3, /*pad_amount=*/3, /*pad_en=*/1);
  };

  float mrow[8], lrow[8];
  v8f acc[8];
#pragma unroll
  for (int r = 0; r < 8; ++r) { mrow[r] = -3.0e38f; lrow[r] = 0.0f; }
#pragma unroll
  for (int nt = 0; nt < 8; ++nt)
#pragma unroll
    for (int j = 0; j < 8; ++j) acc[nt][j] = 0.0f;

  const float scale = 0.08838834764831845f;  // 1/sqrt(128)
  unsigned short* myl = ldsP[wave];

  if (wave == 0) {
    stage_kv(0, 0);
    __builtin_amdgcn_s_wait_tensorcnt(0);
  }
  __syncthreads();

  int p = 0;
  for (int j0 = 0; j0 < SEQ; j0 += 32) {
    // ---- read k fragments from the padded LDS tile ----
    Frag16 kf[8];
#pragma unroll
    for (int f = 0; f < 8; ++f) {
      const unsigned char* kp =
          &ldsK[p][((f >> 2) * 16 + lhalf) * KROW_STRIDE + (f & 3) * 64 + (hi ? 32 : 0)];
      kf[f].u[0] = *(const uint4*)(kp);
      kf[f].u[1] = *(const uint4*)(kp + 16);
    }
    v8f st0, st1;
#pragma unroll
    for (int j = 0; j < 8; ++j) { st0[j] = 0.0f; st1[j] = 0.0f; }
#pragma unroll
    for (int dc = 0; dc < 4; ++dc) {
      st0 = wmma_bf16(qf[dc], kf[dc].v, st0);
      st1 = wmma_bf16(qf[dc], kf[4 + dc].v, st1);
    }

    // ---- kick off next chunk's DMA while this chunk computes ----
    if (j0 + 32 < SEQ && wave == 0) stage_kv(j0 + 32, p ^ 1);

    // ---- read v fragments from the padded LDS tile ----
    Frag16 vf[8];
#pragma unroll
    for (int nt = 0; nt < 8; ++nt) {
      const unsigned char* vp =
          &ldsV[p][(nt * 16 + lhalf) * VROW_STRIDE + (hi ? 32 : 0)];
      vf[nt].u[0] = *(const uint4*)(vp);
      vf[nt].u[1] = *(const uint4*)(vp + 16);
    }

    // ---- online softmax update (rows r / 8+r per lane-half) ----
#pragma unroll
    for (int r = 0; r < 8; ++r) {
      float a0 = st0[r] * scale;
      float a1 = st1[r] * scale;
      float rm = fmaxf(a0, a1);
      rm = fmaxf(rm, __shfl_xor(rm, 1));
      rm = fmaxf(rm, __shfl_xor(rm, 2));
      rm = fmaxf(rm, __shfl_xor(rm, 4));
      rm = fmaxf(rm, __shfl_xor(rm, 8));
      const float mnew = fmaxf(mrow[r], rm);
      const float corr = __expf(mrow[r] - mnew);
      const float p0 = __expf(a0 - mnew);
      const float p1 = __expf(a1 - mnew);
      float rs = p0 + p1;
      rs += __shfl_xor(rs, 1);
      rs += __shfl_xor(rs, 2);
      rs += __shfl_xor(rs, 4);
      rs += __shfl_xor(rs, 8);
      lrow[r] = lrow[r] * corr + rs;
      mrow[r] = mnew;
#pragma unroll
      for (int nt = 0; nt < 8; ++nt) acc[nt][r] *= corr;

      const int prow = hi ? 8 + r : r;
      myl[prow * 32 + lhalf]      = f2bf_bits(p0);
      myl[prow * 32 + 16 + lhalf] = f2bf_bits(p1);
    }

    // Wave-private C->A relayout via LDS; order stores before the reads.
    asm volatile("s_wait_dscnt 0" ::: "memory");
    Frag16 pa;
    {
      const unsigned short* lp = myl + lhalf * 32 + (hi ? 16 : 0);
      pa.u[0] = *(const uint4*)(lp);
      pa.u[1] = *(const uint4*)(lp + 8);
    }
    asm volatile("" ::: "memory");

    // ---- out += p x v for 8 head tiles ----
#pragma unroll
    for (int nt = 0; nt < 8; ++nt) acc[nt] = wmma_bf16(pa.v, vf[nt].v, acc[nt]);

    if (wave == 0) __builtin_amdgcn_s_wait_tensorcnt(0);
    __syncthreads();   // publishes the freshly DMA'd buffer to all waves
    p ^= 1;
  }

  // ---- epilogue: normalize by row sums, store fp32 ----
#pragma unroll
  for (int nt = 0; nt < 8; ++nt) {
    const int col = nt * 16 + lhalf;
#pragma unroll
    for (int r = 0; r < 8; ++r) {
      const int row = mw + (hi ? 8 + r : r);
      out[(size_t)row * DHEAD + col] = acc[nt][r] / lrow[r];
    }
  }
}

// ---------------------------------------------------------------------------
// Host launcher.
// ---------------------------------------------------------------------------
extern "C" void kernel_launch(void* const* d_in, const int* in_sizes, int n_in,
                              void* d_out, int out_size, void* d_ws, size_t ws_size,
                              hipStream_t stream) {
  (void)in_sizes; (void)n_in; (void)out_size; (void)ws_size;

  const float* x  = (const float*)d_in[0];
  const float* Wq = (const float*)d_in[1];
  const float* bq = (const float*)d_in[2];
  const float* Wk = (const float*)d_in[3];
  const float* bk = (const float*)d_in[4];
  const float* Wv = (const float*)d_in[5];
  const float* bv = (const float*)d_in[6];
  float* out = (float*)d_out;

  // Workspace layout (bf16 halves): Wtf | q | k | vT  (~13.5 MB total)
  unsigned short* Wtf = (unsigned short*)d_ws;
  unsigned short* qb  = Wtf + (size_t)3 * DHEAD * DMODEL;
  unsigned short* kbf = qb  + (size_t)BATCH * SEQ * DHEAD;
  unsigned short* vTb = kbf + (size_t)BATCH * SEQ * DHEAD;

  const int ngroups = 3 * NCHUNK * 8 * 2 * 32;
  pack_w_kernel<<<(ngroups + 255) / 256, 256, 0, stream>>>(Wq, Wk, Wv, Wtf);

  dim3 g1(BATCH * SEQ / 128, 3, 1);
  qkv_gemm_kernel<<<g1, 256, 0, stream>>>(x, Wtf, bq, bk, bv, qb, kbf, vTb);

  flash_attn_kernel<<<BATCH * SEQ / 128, 256, 0, stream>>>(qb, kbf, vTb, out);
}
